// selfAttention_8589934592370
// MI455X (gfx1250) — matrix operand order repaired and checked by reading, not compiled, not run on hardware
//
#include <hip/hip_runtime.h>
#include <hip/hip_bf16.h>

// MI455X / gfx1250, wave32. All matmuls via v_wmma_f32_16x16x32_f16.
//
// Transformer-XL relative attention, restructured:
//   logits = [ (q+u)*s | A(q+v,s) ]  @  [ k | B(t) ]^T      (K = 128 f16 GEMM)
// where A/B come from the angle-addition identity for sin/cos((s-t)*w_i),
// eliminating the (d,S,S) R tensor entirely.

typedef __attribute__((ext_vector_type(16))) _Float16 v16h;
typedef __attribute__((ext_vector_type(8)))  _Float16 v8h;
typedef __attribute__((ext_vector_type(8)))  float    v8f;

#define SEQ    1024
#define DMODEL 1024
#define NHEAD  16
#define HDIM   64
#define KAUG   128

static __device__ inline v16h mk16(v8h lo, v8h hi) {
  return __builtin_shufflevector(lo, hi, 0,1,2,3,4,5,6,7,8,9,10,11,12,13,14,15);
}

static __device__ inline float rmax16(float v) {
  v = fmaxf(v, __shfl_xor(v, 1, 16));
  v = fmaxf(v, __shfl_xor(v, 2, 16));
  v = fmaxf(v, __shfl_xor(v, 4, 16));
  v = fmaxf(v, __shfl_xor(v, 8, 16));
  return v;
}
static __device__ inline float rsum16(float v) {
  v += __shfl_xor(v, 1, 16);
  v += __shfl_xor(v, 2, 16);
  v += __shfl_xor(v, 4, 16);
  v += __shfl_xor(v, 8, 16);
  return v;
}

// ---------------- elementwise converts ----------------

__global__ void cvt_f16(const float* __restrict__ x, _Float16* __restrict__ y, int n) {
  int i = blockIdx.x * blockDim.x + threadIdx.x;
  if (i < n) y[i] = (_Float16)x[i];
}

// BT[k*1024 + n] = W[n*1024 + k]   (weight (out,in) -> K-major f16 for GEMM B)
__global__ void transpose_cvt(const float* __restrict__ W, _Float16* __restrict__ BT) {
  int i = blockIdx.x * blockDim.x + threadIdx.x;
  int k = i >> 10, n = i & 1023;
  BT[i] = (_Float16)W[n * DMODEL + k];
}

// ---------------- 1024^3 WMMA GEMM: C = A(f16,MxK) * B(f16,KxN) [+ bias] ----------------
// block = 4 waves; block tile 64x64; K-step 32; A row-major + B transposed in LDS
// so every fragment read is a contiguous 16B ds_load.

__global__ __launch_bounds__(128) void gemm1024_f16(
    const _Float16* __restrict__ A, const _Float16* __restrict__ B,
    float* __restrict__ C, const float* __restrict__ bias) {
  __shared__ alignas(16) _Float16 As[64 * 32];
  __shared__ alignas(16) _Float16 BsT[64 * 32];

  const int lane = threadIdx.x & 31;
  const int wave = threadIdx.x >> 5;
  const int m0 = blockIdx.y * 64;
  const int n0 = blockIdx.x * 64;
  const int nf   = lane & 15;
  const int kbA  = (lane < 16) ? 0 : 8;   // A frag K base (16-bit A layout)
  const int kbB  = (lane < 16) ? 0 : 8;   // B frag K base (16-bit B layout)

  v8f acc[4];
  #pragma unroll
  for (int nt = 0; nt < 4; ++nt)
    #pragma unroll
    for (int r = 0; r < 8; ++r) acc[nt][r] = 0.0f;

  for (int k0 = 0; k0 < 1024; k0 += 32) {
    { // stage A tile 64x32 (row-major)
      int row = threadIdx.x >> 1;
      int cb  = (threadIdx.x & 1) * 16;
      const _Float16* src = A + (size_t)(m0 + row) * 1024 + k0 + cb;
      _Float16* dst = As + row * 32 + cb;
      *(v8h*)dst       = *(const v8h*)src;
      *(v8h*)(dst + 8) = *(const v8h*)(src + 8);
    }
    { // stage B tile 32x64, transposed -> BsT[n][k]
      int krow = threadIdx.x >> 2;
      int nb   = (threadIdx.x & 3) * 16;
      const _Float16* src = B + (size_t)(k0 + krow) * 1024 + n0 + nb;
      #pragma unroll
      for (int i = 0; i < 16; ++i) BsT[(nb + i) * 32 + krow] = src[i];
    }
    __syncthreads();

    const _Float16* arow = As + (wave * 16 + nf) * 32 + kbA;
    v16h a = mk16(*(const v8h*)arow, *(const v8h*)(arow + 16));
    #pragma unroll
    for (int nt = 0; nt < 4; ++nt) {
      const _Float16* brow = BsT + (nt * 16 + nf) * 32 + kbB;
      v16h b = mk16(*(const v8h*)brow, *(const v8h*)(brow + 16));
      acc[nt] = __builtin_amdgcn_wmma_f32_16x16x32_f16(false, a, false, b,
                                                       (short)0, acc[nt], false, false);
    }
    __syncthreads();
  }

  const int rowoff = (lane >> 4) << 3;
  #pragma unroll
  for (int nt = 0; nt < 4; ++nt) {
    const int col = n0 + nt * 16 + nf;
    const float badd = bias ? bias[col] : 0.0f;  // uniform branch, 1 load per N-tile
    #pragma unroll
    for (int r = 0; r < 8; ++r) {
      int row = m0 + wave * 16 + r + rowoff;
      C[(size_t)row * 1024 + col] = acc[nt][r] + badd;
    }
  }
}

// ---------------- build augmented attention operands ----------------
// QU[h,s,0:64]  = (q + u_h) * d^-1/2          KB[h,t,0:64]  = k
// QU[h,s,64+2i] = qv[2i]*sin(sw)+qv[2i+1]*cos(sw)   KB[h,t,64+2i]   = cos(tw)
// QU[h,s,65+2i] =-qv[2i]*cos(sw)+qv[2i+1]*sin(sw)   KB[h,t,65+2i]   = sin(tw)
// => QU[h,s,:] . KB[h,t,:] == (cc + cp + uk + vr)[h,s,t] * d^-1/2

__global__ void build_attin(const float* __restrict__ Qf, const float* __restrict__ Kf,
                            const float* __restrict__ Vf, const float* __restrict__ ub,
                            const float* __restrict__ vb,
                            _Float16* __restrict__ QU, _Float16* __restrict__ KB,
                            _Float16* __restrict__ Vh) {
  int id = blockIdx.x * blockDim.x + threadIdx.x;  // h*SEQ + s
  if (id >= NHEAD * SEQ) return;
  int h = id >> 10;
  int s = id & 1023;
  const float scale = 0.125f;  // 64^-0.5

  float qv[64];
  const float* qrow = Qf + (size_t)s * 1024 + h * 64;
  const float* krow = Kf + (size_t)s * 1024 + h * 64;
  const float* vrow = Vf + (size_t)s * 1024 + h * 64;
  _Float16* quo = QU + (size_t)id * KAUG;
  _Float16* kbo = KB + (size_t)id * KAUG;
  _Float16* vho = Vh + (size_t)id * HDIM;

  #pragma unroll 8
  for (int dd = 0; dd < 64; ++dd) {
    float q = qrow[dd];
    quo[dd] = (_Float16)((q + ub[h * 64 + dd]) * scale);
    qv[dd]  = (q + vb[h * 64 + dd]) * scale;
    kbo[dd] = (_Float16)krow[dd];
    vho[dd] = (_Float16)vrow[dd];
  }
  const float c = 13.287712379549449f / 32.0f;  // log2(10000)/32
  for (int i = 0; i < 32; ++i) {
    float w   = exp2f(-(float)i * c);           // 10000^(-2i/64)
    float ang = (float)s * w;
    float sw = __sinf(ang), cw = __cosf(ang);
    quo[64 + 2 * i]     = (_Float16)( qv[2 * i] * sw + qv[2 * i + 1] * cw);
    quo[64 + 2 * i + 1] = (_Float16)(-qv[2 * i] * cw + qv[2 * i + 1] * sw);
    kbo[64 + 2 * i]     = (_Float16)cw;
    kbo[64 + 2 * i + 1] = (_Float16)sw;
  }
}

// ---------------- attention: 1 wave per (head, 16 query rows) ----------------
// pass 1: logits (4 WMMAs / 16-key tile, K=128) + shfl row max/sum
// pass 2: recompute logits, P=exp(.) -> f16 LDS, P@V WMMAs (K=32 key pairs)

__global__ __launch_bounds__(32) void attn_f16(
    const _Float16* __restrict__ QU, const _Float16* __restrict__ KB,
    const _Float16* __restrict__ Vh, _Float16* __restrict__ O) {
  __shared__ alignas(16) _Float16 Pt[16 * 32];
  __shared__ alignas(16) _Float16 VtT[64 * 32];

  const int lane = threadIdx.x;
  const int h  = blockIdx.y;
  const int s0 = blockIdx.x * 16;
  const int nf  = lane & 15;
  const int kbA = (lane < 16) ? 0 : 8;
  const int kbB = (lane < 16) ? 0 : 8;

  // A fragments of QU row block (16 x 128) -> 4 frags, kept in VGPRs all kernel
  v16h aq[4];
  {
    const _Float16* arow = QU + (size_t)((h << 10) + s0 + nf) * KAUG + kbA;
    #pragma unroll
    for (int kk = 0; kk < 4; ++kk)
      aq[kk] = mk16(*(const v8h*)(arow + kk * 32), *(const v8h*)(arow + kk * 32 + 16));
  }
  const _Float16* KBh = KB + (size_t)(h << 10) * KAUG;
  const _Float16* Vhh = Vh + (size_t)(h << 10) * HDIM;

  float runmax[8], runsum[8];
  #pragma unroll
  for (int r = 0; r < 8; ++r) { runmax[r] = -1e30f; runsum[r] = 0.0f; }

  // ---- pass 1: softmax statistics ----
  for (int t0 = 0; t0 < SEQ; t0 += 16) {
    v8f acc = {0.f, 0.f, 0.f, 0.f, 0.f, 0.f, 0.f, 0.f};
    const _Float16* brow = KBh + (size_t)(t0 + nf) * KAUG + kbB;
    #pragma unroll
    for (int kk = 0; kk < 4; ++kk) {
      v16h b = mk16(*(const v8h*)(brow + kk * 32), *(const v8h*)(brow + kk * 32 + 16));
      acc = __builtin_amdgcn_wmma_f32_16x16x32_f16(false, aq[kk], false, b,
                                                   (short)0, acc, false, false);
    }
    #pragma unroll
    for (int r = 0; r < 8; ++r) {
      float x  = acc[r];
      float nm = fmaxf(runmax[r], rmax16(x));
      float es = rsum16(__expf(x - nm));
      runsum[r] = runsum[r] * __expf(runmax[r] - nm) + es;
      runmax[r] = nm;
    }
  }

  // ---- pass 2: P @ V ----
  v8f oacc[4];
  #pragma unroll
  for (int nt = 0; nt < 4; ++nt)
    #pragma unroll
    for (int r = 0; r < 8; ++r) oacc[nt][r] = 0.0f;

  const int rowoff = (lane >> 4) << 3;
  for (int t0 = 0; t0 < SEQ; t0 += 32) {
    { // stage V tile transposed: VtT[dd][k] = V[t0+k][dd]
      const _Float16* vsrc = Vhh + (size_t)(t0 + lane) * HDIM;
      #pragma unroll
      for (int dd = 0; dd < 64; ++dd) VtT[dd * 32 + lane] = vsrc[dd];
    }
    #pragma unroll
    for (int half = 0; half < 2; ++half) {
      int tt = t0 + half * 16;
      v8f acc = {0.f, 0.f, 0.f, 0.f, 0.f, 0.f, 0.f, 0.f};
      const _Float16* brow = KBh + (size_t)(tt + nf) * KAUG + kbB;
      #pragma unroll
      for (int kk = 0; kk < 4; ++kk) {
        v16h b = mk16(*(const v8h*)(brow + kk * 32), *(const v8h*)(brow + kk * 32 + 16));
        acc = __builtin_amdgcn_wmma_f32_16x16x32_f16(false, aq[kk], false, b,
                                                     (short)0, acc, false, false);
      }
      #pragma unroll
      for (int r = 0; r < 8; ++r) {
        float p = __expf(acc[r] - runmax[r]);
        Pt[(r + rowoff) * 32 + half * 16 + nf] = (_Float16)p;
      }
    }
    __syncthreads();  // 1-wave WG: S_NOP, but orders LDS traffic

    {
      const _Float16* arow = Pt + nf * 32 + kbA;
      v16h ap = mk16(*(const v8h*)arow, *(const v8h*)(arow + 16));
      #pragma unroll
      for (int nt = 0; nt < 4; ++nt) {
        const _Float16* brow2 = VtT + (nt * 16 + nf) * 32 + kbB;
        v16h bv = mk16(*(const v8h*)brow2, *(const v8h*)(brow2 + 16));
        oacc[nt] = __builtin_amdgcn_wmma_f32_16x16x32_f16(false, ap, false, bv,
                                                          (short)0, oacc[nt], false, false);
      }
    }
    __syncthreads();
  }

  // store O[s, h*64 + dd] (concat-heads layout), normalized by row sums
  #pragma unroll
  for (int nt = 0; nt < 4; ++nt)
    #pragma unroll
    for (int r = 0; r < 8; ++r) {
      int srow = s0 + r + rowoff;
      int col  = (h << 6) + nt * 16 + nf;
      O[(size_t)srow * 1024 + col] = (_Float16)(oacc[nt][r] / runsum[r]);
    }
}

// ---------------- host ----------------

extern "C" void kernel_launch(void* const* d_in, const int* in_sizes, int n_in,
                              void* d_out, int out_size, void* d_ws, size_t ws_size,
                              hipStream_t stream) {
  const float* val  = (const float*)d_in[0];
  const float* keyx = (const float*)d_in[1];
  const float* qry  = (const float*)d_in[2];
  const float* Wq   = (const float*)d_in[3];
  const float* Wk   = (const float*)d_in[4];
  const float* Wv   = (const float*)d_in[5];
  const float* ub   = (const float*)d_in[6];
  const float* vb   = (const float*)d_in[7];
  const float* Wo   = (const float*)d_in[8];
  const float* bo   = (const float*)d_in[9];
  float* out = (float*)d_out;

  char* ws = (char*)d_ws;
  size_t off = 0;
  auto alloc = [&](size_t bytes) -> char* {
    char* p = ws + off;
    off += (bytes + 255) & ~(size_t)255;
    return p;
  };

  const size_t M1 = (size_t)1024 * 1024;
  _Float16* qh  = (_Float16*)alloc(M1 * 2);
  _Float16* kh  = (_Float16*)alloc(M1 * 2);
  _Float16* vh  = (_Float16*)alloc(M1 * 2);
  _Float16* wqT = (_Float16*)alloc(M1 * 2);
  _Float16* wkT = (_Float16*)alloc(M1 * 2);
  _Float16* wvT = (_Float16*)alloc(M1 * 2);
  _Float16* woT = (_Float16*)alloc(M1 * 2);
  float* Qf = (float*)alloc(M1 * 4);
  float* Kf = (float*)alloc(M1 * 4);
  float* Vf = (float*)alloc(M1 * 4);
  _Float16* QUb = (_Float16*)alloc((size_t)NHEAD * SEQ * KAUG * 2);
  _Float16* KBb = (_Float16*)alloc((size_t)NHEAD * SEQ * KAUG * 2);
  _Float16* Vhb = (_Float16*)alloc((size_t)NHEAD * SEQ * HDIM * 2);
  _Float16* Ob  = (_Float16*)alloc(M1 * 2);

  const int N1 = (int)M1;
  cvt_f16<<<N1 / 256, 256, 0, stream>>>(qry,  qh, N1);
  cvt_f16<<<N1 / 256, 256, 0, stream>>>(keyx, kh, N1);
  cvt_f16<<<N1 / 256, 256, 0, stream>>>(val,  vh, N1);
  transpose_cvt<<<N1 / 256, 256, 0, stream>>>(Wq, wqT);
  transpose_cvt<<<N1 / 256, 256, 0, stream>>>(Wk, wkT);
  transpose_cvt<<<N1 / 256, 256, 0, stream>>>(Wv, wvT);
  transpose_cvt<<<N1 / 256, 256, 0, stream>>>(Wo, woT);

  dim3 gg(16, 16);
  gemm1024_f16<<<gg, 128, 0, stream>>>(qh, wqT, Qf, nullptr);
  gemm1024_f16<<<gg, 128, 0, stream>>>(kh, wkT, Kf, nullptr);
  gemm1024_f16<<<gg, 128, 0, stream>>>(vh, wvT, Vf, nullptr);

  build_attin<<<(NHEAD * SEQ) / 128, 128, 0, stream>>>(Qf, Kf, Vf, ub, vb, QUb, KBb, Vhb);

  attn_f16<<<dim3(SEQ / 16, NHEAD), 32, 0, stream>>>(QUb, KBb, Vhb, Ob);

  gemm1024_f16<<<gg, 128, 0, stream>>>(Ob, woT, out, bo);
}
